// MODEL_12472585027859
// MI455X (gfx1250) — compile-verified, matrix operand order checked by reading, not run on hardware
//
#include <hip/hip_runtime.h>
#include <hip/hip_bf16.h>

// ---------------------------------------------------------------------------
// Problem constants (match reference)
// ---------------------------------------------------------------------------
#define NU    100000
#define NI    25000
#define NALL  (NU + NI)
#define DD    64
#define KK    3
#define DK    (DD * KK)   // 192
#define E_UU  1000000
#define E_II  500000
#define E_UI  1000000
#define LLAY  2
#define ALPHA 0.25f

typedef __attribute__((ext_vector_type(16))) _Float16 v16h;
typedef __attribute__((ext_vector_type(8)))  _Float16 v8h;
typedef __attribute__((ext_vector_type(8)))  float    v8f;

// ---------------------------------------------------------------------------
// WMMA fragment loaders (CDNA5 16-bit layouts, wave32).
// Per frag a lane owns K = [k0+half*8, +8) and [k0+16+half*8, +8): two
// contiguous chunks. f16 source -> two global_load_b128; f32 -> v8f + cvt.
// ---------------------------------------------------------------------------
__device__ __forceinline__ v16h load_frag(const _Float16* __restrict__ base,
                                          int stride, int r, int k0, int half) {
  const _Float16* p = base + (size_t)r * stride + k0 + half * 8;
  v8h c0 = *(const v8h*)(p);
  v8h c1 = *(const v8h*)(p + 16);
  return __builtin_shufflevector(c0, c1, 0, 1, 2, 3, 4, 5, 6, 7,
                                 8, 9, 10, 11, 12, 13, 14, 15);
}

__device__ __forceinline__ v16h load_frag(const float* __restrict__ base,
                                          int stride, int r, int k0, int half) {
  const float* p = base + (size_t)r * stride + k0 + half * 8;
  v8f c0 = *(const v8f*)(p);
  v8f c1 = *(const v8f*)(p + 16);
  v16h out;
#pragma unroll
  for (int i = 0; i < 8; ++i) out[i] = (_Float16)c0[i];
#pragma unroll
  for (int i = 0; i < 8; ++i) out[8 + i] = (_Float16)c1[i];
  return out;
}

__device__ __forceinline__ float sigmoidf_(float x) {
  return __builtin_amdgcn_rcpf(1.0f + __expf(-x));
}

// ---------------------------------------------------------------------------
// Generic WMMA GEMM: C = act(A @ B + bias)
//   A: MxKD (f32 or f16) row-major; BT: NxKD f16 row-major (pre-transposed B)
//   act: 0 = identity, 1 = leaky-relu(ALPHA), 2 = gmul * sigmoid(.)
// One wave computes a 16x64 strip (4 col tiles): each A frag loaded once and
// reused across 4 WMMAs -> 4x less A traffic on the N=192 GEMMs.
// ---------------------------------------------------------------------------
template <int KD, typename TA, typename TO>
__global__ __launch_bounds__(256)
void wmma_gemm_t(const TA* __restrict__ A, int lda,
                 const _Float16* __restrict__ BT,
                 const float* __restrict__ bias,
                 const float* __restrict__ gmul,
                 TO* __restrict__ C, int ldc,
                 int M, int N, int act) {
  int gwave = (int)((blockIdx.x * blockDim.x + threadIdx.x) >> 5);
  int lane  = threadIdx.x & 31;
  int strips = N >> 6;                   // N / 64
  int tilesM = (M + 15) >> 4;
  if (gwave >= tilesM * strips) return;
  int tm = gwave / strips, ts = gwave % strips;
  int row0 = tm << 4, col0 = ts << 6;
  int half = lane >> 4;
  int arow = row0 + (lane & 15);
  if (arow >= M) arow = M - 1;           // clamp partial tile; EXEC stays full
  int bcol = lane & 15;

  v8f c[4];
#pragma unroll
  for (int t = 0; t < 4; ++t) c[t] = (v8f){0.f, 0.f, 0.f, 0.f, 0.f, 0.f, 0.f, 0.f};

#pragma unroll
  for (int k0 = 0; k0 < KD; k0 += 32) {
    v16h a = load_frag(A, lda, arow, k0, half);
#pragma unroll
    for (int t = 0; t < 4; ++t) {
      v16h b = load_frag(BT, KD, col0 + t * 16 + bcol, k0, half);
      c[t] = __builtin_amdgcn_wmma_f32_16x16x32_f16(false, a, false, b, (short)0, c[t], false, false);
    }
  }

#pragma unroll
  for (int t = 0; t < 4; ++t) {
    int col = col0 + t * 16 + bcol;
    float bv = bias ? bias[col] : 0.0f;
#pragma unroll
    for (int v = 0; v < 8; ++v) {
      int row = row0 + v + (half << 3);
      if (row < M) {
        float x = c[t][v] + bv;
        if (act == 1)      x = (x > 0.0f) ? x : ALPHA * x;
        else if (act == 2) x = gmul[(size_t)row * ldc + col] * sigmoidf_(x);
        C[(size_t)row * ldc + col] = (TO)x;
      }
    }
  }
}

// ---------------------------------------------------------------------------
// WMMA GEMM with A = concat([A0|A1|A2], axis=1), each 64 wide (f32). K=192, N=64.
// C = A @ B + bias (tu / ti projection); BT: 64x192 f16. 16x64 strip per wave.
// ---------------------------------------------------------------------------
__global__ __launch_bounds__(256)
void wmma_gemm_cat3_kernel(const float* __restrict__ A0, const float* __restrict__ A1,
                           const float* __restrict__ A2,
                           const _Float16* __restrict__ BT, const float* __restrict__ bias,
                           float* __restrict__ C, int M) {
  int gwave = (int)((blockIdx.x * blockDim.x + threadIdx.x) >> 5);
  int lane  = threadIdx.x & 31;
  int tilesM = (M + 15) >> 4;
  if (gwave >= tilesM) return;
  int row0 = gwave << 4;
  int half = lane >> 4;
  int arow = row0 + (lane & 15);
  if (arow >= M) arow = M - 1;
  int bcol = lane & 15;

  const float* parts[3] = {A0, A1, A2};
  v8f c[4];
#pragma unroll
  for (int t = 0; t < 4; ++t) c[t] = (v8f){0.f, 0.f, 0.f, 0.f, 0.f, 0.f, 0.f, 0.f};

#pragma unroll
  for (int k0 = 0; k0 < DK; k0 += 32) {  // each 32-chunk stays inside one part
    v16h a = load_frag(parts[k0 / 64], DD, arow, k0 & 63, half);
#pragma unroll
    for (int t = 0; t < 4; ++t) {
      v16h b = load_frag(BT, DK, t * 16 + bcol, k0, half);
      c[t] = __builtin_amdgcn_wmma_f32_16x16x32_f16(false, a, false, b, (short)0, c[t], false, false);
    }
  }

#pragma unroll
  for (int t = 0; t < 4; ++t) {
    int col = t * 16 + bcol;
    float bv = bias ? bias[col] : 0.0f;
#pragma unroll
    for (int v = 0; v < 8; ++v) {
      int row = row0 + v + (half << 3);
      if (row < M) C[(size_t)row * DD + col] = c[t][v] + bv;
    }
  }
}

// ---------------------------------------------------------------------------
// Weight prep: B (KxN f32 row-major) -> BT (NxK f16 row-major)
// ---------------------------------------------------------------------------
__global__ void transpose_f16_kernel(const float* __restrict__ B, _Float16* __restrict__ BT,
                                     int Kd, int N) {
  int t = blockIdx.x * blockDim.x + threadIdx.x;
  if (t >= Kd * N) return;
  int k = t / N, n = t % N;
  BT[(size_t)n * Kd + k] = (_Float16)B[t];
}

// ---------------------------------------------------------------------------
// Graph-normalization kernels
// ---------------------------------------------------------------------------
__global__ void deg_count_kernel(const int* __restrict__ row, float* __restrict__ deg,
                                 int nE, int off) {
  int t = blockIdx.x * blockDim.x + threadIdx.x;
  if (t < nE) atomicAdd(&deg[row[t] + off], 1.0f);
}

__global__ void deg_to_dinv_kernel(float* __restrict__ deg, int n) {
  int t = blockIdx.x * blockDim.x + threadIdx.x;
  if (t < n) { float d = deg[t]; deg[t] = (d > 0.0f) ? rsqrtf(d) : 0.0f; }
}

__global__ void edge_vals_kernel(const int* __restrict__ r, const int* __restrict__ c,
                                 const float* __restrict__ dinv, float* __restrict__ val,
                                 int nE, int roff, int coff) {
  int t = blockIdx.x * blockDim.x + threadIdx.x;
  if (t < nE) val[t] = dinv[r[t] + roff] * dinv[c[t] + coff];
}

// ---------------------------------------------------------------------------
// SpMM: out[row[e]] += val[e] * x[col[e]]   (wave per edge, float2 per lane)
// ---------------------------------------------------------------------------
__global__ __launch_bounds__(256)
void spmm_atomic_kernel(const int* __restrict__ row, const int* __restrict__ col,
                        const float* __restrict__ val,
                        const float* __restrict__ x, float* __restrict__ out, int nE) {
  int wave = (int)((blockIdx.x * blockDim.x + threadIdx.x) >> 5);
  int lane = threadIdx.x & 31;
  if (wave >= nE) return;
  int r = row[wave], c = col[wave];
  float v = val ? val[wave] : 1.0f;
  float2 d = ((const float2*)(x + (size_t)c * DD))[lane];
  float* o = out + (size_t)r * DD + lane * 2;
  atomicAdd(o,     v * d.x);
  atomicAdd(o + 1, v * d.y);
}

// Bidirectional user<->item SpMM in the concatenated N_ALL space.
__global__ __launch_bounds__(256)
void spmm_ui_kernel(const int* __restrict__ uu, const int* __restrict__ ii,
                    const float* __restrict__ val,
                    const float* __restrict__ xall, float* __restrict__ outall, int nE) {
  int wave = (int)((blockIdx.x * blockDim.x + threadIdx.x) >> 5);
  int lane = threadIdx.x & 31;
  if (wave >= nE) return;
  int u = uu[wave];
  int it = ii[wave] + NU;
  float v = val[wave];
  float2 xu = ((const float2*)(xall + (size_t)u  * DD))[lane];
  float2 xi = ((const float2*)(xall + (size_t)it * DD))[lane];
  float* ou = outall + (size_t)u  * DD + lane * 2;
  float* oi = outall + (size_t)it * DD + lane * 2;
  atomicAdd(ou,     v * xi.x);
  atomicAdd(ou + 1, v * xi.y);
  atomicAdd(oi,     v * xu.x);
  atomicAdd(oi + 1, v * xu.y);
}

// ---------------------------------------------------------------------------
// Layer combine: cur = 0.5*(seg + uiv); acc += l2norm(seg); accUI += l2norm(uiv)
// ---------------------------------------------------------------------------
__global__ __launch_bounds__(256)
void layer_combine_kernel(const float* __restrict__ u0, const float* __restrict__ i0,
                          const float* __restrict__ uiv,
                          float* __restrict__ curU, float* __restrict__ curI,
                          float* __restrict__ curUI,
                          float* __restrict__ accU, float* __restrict__ accI,
                          float* __restrict__ accUI) {
  int wave = (int)((blockIdx.x * blockDim.x + threadIdx.x) >> 5);
  int lane = threadIdx.x & 31;
  if (wave >= NALL) return;
  bool isU = wave < NU;
  size_t r64 = (size_t)wave * DD;
  size_t s64 = isU ? r64 : (size_t)(wave - NU) * DD;
  const float* arow = isU ? (u0 + s64) : (i0 + s64);
  const float* brow = uiv + r64;
  int j = lane * 2;
  float a0 = arow[j], a1 = arow[j + 1];
  float b0 = brow[j], b1 = brow[j + 1];
  float ssa = a0 * a0 + a1 * a1;
  float ssb = b0 * b0 + b1 * b1;
#pragma unroll
  for (int off = 16; off; off >>= 1) {
    ssa += __shfl_xor(ssa, off, 32);
    ssb += __shfl_xor(ssb, off, 32);
  }
  float sa = __builtin_amdgcn_rcpf(fmaxf(sqrtf(ssa), 1e-12f));
  float sb = __builtin_amdgcn_rcpf(fmaxf(sqrtf(ssb), 1e-12f));
  float* acc = isU ? (accU + s64) : (accI + s64);
  acc[j]     += a0 * sa;
  acc[j + 1] += a1 * sa;
  accUI[r64 + j]     += b0 * sb;
  accUI[r64 + j + 1] += b1 * sb;
  float n0 = 0.5f * (a0 + b0), n1 = 0.5f * (a1 + b1);
  curUI[r64 + j] = n0;  curUI[r64 + j + 1] = n1;
  float* cur = isU ? (curU + s64) : (curI + s64);
  cur[j] = n0;  cur[j + 1] = n1;
}

__global__ void scale_inplace_kernel(float* __restrict__ x, int n, float s) {
  int t = blockIdx.x * blockDim.x + threadIdx.x;
  if (t < n) x[t] *= s;
}

// uiE mean -> split directly into d_out sections (ui_uE, ui_iE)
__global__ void finalize_ui_kernel(const float* __restrict__ accUI,
                                   float* __restrict__ outUiU, float* __restrict__ outUiI) {
  int t = blockIdx.x * blockDim.x + threadIdx.x;
  if (t >= NALL * DD) return;
  float v = accUI[t] * (1.0f / (LLAY + 1));
  int row = t >> 6, col = t & 63;
  if (row < NU) outUiU[(row << 6) + col] = v;
  else          outUiI[((row - NU) << 6) + col] = v;
}

// ---------------------------------------------------------------------------
// Row l2norm (width w), one wave per row.
// ---------------------------------------------------------------------------
__global__ __launch_bounds__(256)
void l2norm_rows_kernel(const float* __restrict__ in, float* __restrict__ out, int n, int w) {
  int wave = (int)((blockIdx.x * blockDim.x + threadIdx.x) >> 5);
  int lane = threadIdx.x & 31;
  if (wave >= n) return;
  const float* r = in + (size_t)wave * w;
  float ss = 0.0f;
  for (int j = lane; j < w; j += 32) { float v = r[j]; ss += v * v; }
#pragma unroll
  for (int off = 16; off; off >>= 1) ss += __shfl_xor(ss, off, 32);
  float s = __builtin_amdgcn_rcpf(fmaxf(sqrtf(ss), 1e-12f));
  float* o = out + (size_t)wave * w;
  for (int j = lane; j < w; j += 32) o[j] = r[j] * s;
}

// Column means of (n x 192)
__global__ void col_mean_kernel(const float* __restrict__ m, int n, float* __restrict__ mean) {
  __shared__ float sh[256];
  int col = blockIdx.x;
  float s = 0.0f;
  for (int r = threadIdx.x; r < n; r += blockDim.x) s += m[(size_t)r * DK + col];
  sh[threadIdx.x] = s;
  __syncthreads();
  for (int off = 128; off; off >>= 1) {
    if ((int)threadIdx.x < off) sh[threadIdx.x] += sh[threadIdx.x + off];
    __syncthreads();
  }
  if (threadIdx.x == 0) mean[col] = sh[0] * __builtin_amdgcn_rcpf((float)n);
}

// softmax over D axis of m1 (n,D,K) + dot with E -> sbuf[n,K]
__global__ void softmax_d_dot_kernel(const float* __restrict__ m1, const float* __restrict__ mean1,
                                     const float* __restrict__ E, float* __restrict__ sbuf, int n) {
  int t = blockIdx.x * blockDim.x + threadIdx.x;
  if (t >= n * KK) return;
  int nn = t / KK, k = t % KK;
  const float* row = m1 + (size_t)nn * DK;
  float mx = -3.4e38f;
  for (int d = 0; d < DD; ++d) mx = fmaxf(mx, row[d * KK + k] + mean1[d * KK + k]);
  float s = 0.0f;
  for (int d = 0; d < DD; ++d) s += __expf(row[d * KK + k] + mean1[d * KK + k] - mx);
  float inv = __builtin_amdgcn_rcpf(s);
  const float* e = E + (size_t)nn * DD;
  float acc = 0.0f;
  for (int d = 0; d < DD; ++d)
    acc += e[d] * __expf(row[d * KK + k] + mean1[d * KK + k] - mx) * inv;
  sbuf[t] = acc;
}

// softmax over K axis of m2 (n,K,D), combine with sbuf, add E, write output.
__global__ void softmax_k_out_kernel(const float* __restrict__ m2, const float* __restrict__ mean2,
                                     const float* __restrict__ sbuf, const float* __restrict__ E,
                                     float* __restrict__ outp, int n) {
  int t = blockIdx.x * blockDim.x + threadIdx.x;
  if (t >= n * DD) return;
  int nn = t >> 6, d = t & 63;
  const float* row = m2 + (size_t)nn * DK;
  float v0 = row[0 * DD + d] + mean2[0 * DD + d];
  float v1 = row[1 * DD + d] + mean2[1 * DD + d];
  float v2 = row[2 * DD + d] + mean2[2 * DD + d];
  float mx = fmaxf(v0, fmaxf(v1, v2));
  float e0 = __expf(v0 - mx), e1 = __expf(v1 - mx), e2 = __expf(v2 - mx);
  float inv = __builtin_amdgcn_rcpf(e0 + e1 + e2);
  const float* s = sbuf + (size_t)nn * KK;
  float r = (s[0] * e0 + s[1] * e1 + s[2] * e2) * inv;
  outp[t] = E[t] + r;
}

// ---------------------------------------------------------------------------
// Host-side orchestration
// ---------------------------------------------------------------------------
static inline int cdiv(int a, int b) { return (a + b - 1) / b; }
// grid for strip GEMMs: one wave per (16-row block x 64-col strip)
static inline int gemm_blocks(int M, int N) { return cdiv(((M + 15) / 16) * (N / 64), 8); }

extern "C" void kernel_launch(void* const* d_in, const int* in_sizes, int n_in,
                              void* d_out, int out_size, void* d_ws, size_t ws_size,
                              hipStream_t stream) {
  (void)in_sizes; (void)n_in; (void)out_size; (void)ws_size;

  // Inputs (setup_inputs dict order)
  const float* user_emb = (const float*)d_in[0];
  const float* item_emb = (const float*)d_in[1];
  const float* gwu  = (const float*)d_in[2];
  const float* gwub = (const float*)d_in[3];
  const float* gwi  = (const float*)d_in[4];
  const float* gwib = (const float*)d_in[5];
  const float* mnu_W = (const float*)d_in[6];
  const float* mnu_b = (const float*)d_in[7];
  const float* mni_W = (const float*)d_in[8];
  const float* mni_b = (const float*)d_in[9];
  const float* mlpW[4][4]; // [idx][Wp,bp,Wo,bo]
  for (int i = 0; i < 4; ++i)
    for (int j = 0; j < 4; ++j)
      mlpW[i][j] = (const float*)d_in[10 + i * 4 + j];
  const int* uu_src = (const int*)d_in[26];
  const int* uu_dst = (const int*)d_in[27];
  const int* ii_src = (const int*)d_in[28];
  const int* ii_dst = (const int*)d_in[29];
  const int* ui_u   = (const int*)d_in[30];
  const int* ui_i   = (const int*)d_in[31];

  float* out = (float*)d_out;
  const size_t oUser = 0;
  const size_t oItem = (size_t)NU * DD;
  const size_t oUiU  = oItem + (size_t)NI * DD;
  const size_t oUiI  = oUiU + (size_t)NU * DD;

  // Workspace layout (f32 region, then f16 region; offsets kept 32B-aligned)
  float* ws = (float*)d_ws;
  size_t off = 0;
  auto alloc = [&](size_t n) { float* p = ws + off; off += (n + 7) & ~(size_t)7; return p; };
  float* uuVal = alloc(E_UU);
  float* iiVal = alloc(E_II);
  float* uiVal = alloc(E_UI);
  float* deg   = alloc(NALL);
  float* accU  = alloc((size_t)NU * DD);     // becomes userE
  float* accI  = alloc((size_t)NI * DD);     // becomes itemE
  float* accUI = alloc((size_t)NALL * DD);
  float* curU  = alloc((size_t)NU * DD);
  float* curI  = alloc((size_t)NI * DD);
  float* curUI = alloc((size_t)NALL * DD);
  float* tmpU  = alloc((size_t)NU * DD);     // u0 / uneigh
  float* tmpI  = alloc((size_t)NI * DD);     // i0 / ineigh
  float* tmpUI = alloc((size_t)NALL * DD);   // uiv
  float* tu    = alloc((size_t)NU * DD);
  float* ti    = alloc((size_t)NI * DD);
  float* m1    = alloc((size_t)NU * DK);
  float* m2    = alloc((size_t)NU * DK);
  float* mean1 = alloc(DK);
  float* mean2 = alloc(DK);
  float* sbuf  = alloc((size_t)NU * KK);

  // f16 region
  _Float16* wh = (_Float16*)alloc(0);
  size_t hoff = 0;
  auto halloc = [&](size_t n) { _Float16* p = wh + hoff; hoff += (n + 15) & ~(size_t)15; return p; };
  _Float16* hbuf = halloc((size_t)NU * DK);      // f16 hidden activation
  _Float16* gwuT = halloc(DD * DD);              // 64 x 64   (N x K)
  _Float16* gwiT = halloc(DD * DD);
  _Float16* mnuT = halloc(DD * DK);              // 64 x 192
  _Float16* mniT = halloc(DD * DK);
  _Float16* WpT[4], *WoT[4];
  for (int i = 0; i < 4; ++i) {
    WpT[i] = halloc(DK * DD);                    // 192 x 64
    WoT[i] = halloc(DK * DK);                    // 192 x 192
  }

  const int T = 256;

  // ---- 0. convert/transpose weights to f16 (NxK) ------------------------
  transpose_f16_kernel<<<cdiv(DD * DD, T), T, 0, stream>>>(gwu, gwuT, DD, DD);
  transpose_f16_kernel<<<cdiv(DD * DD, T), T, 0, stream>>>(gwi, gwiT, DD, DD);
  transpose_f16_kernel<<<cdiv(DK * DD, T), T, 0, stream>>>(mnu_W, mnuT, DK, DD);
  transpose_f16_kernel<<<cdiv(DK * DD, T), T, 0, stream>>>(mni_W, mniT, DK, DD);
  for (int i = 0; i < 4; ++i) {
    transpose_f16_kernel<<<cdiv(DD * DK, T), T, 0, stream>>>(mlpW[i][0], WpT[i], DD, DK);
    transpose_f16_kernel<<<cdiv(DK * DK, T), T, 0, stream>>>(mlpW[i][2], WoT[i], DK, DK);
  }

  // ---- 1. symmetric normalization values --------------------------------
  hipMemsetAsync(deg, 0, sizeof(float) * NU, stream);
  deg_count_kernel<<<cdiv(E_UU, T), T, 0, stream>>>(uu_src, deg, E_UU, 0);
  deg_to_dinv_kernel<<<cdiv(NU, T), T, 0, stream>>>(deg, NU);
  edge_vals_kernel<<<cdiv(E_UU, T), T, 0, stream>>>(uu_src, uu_dst, deg, uuVal, E_UU, 0, 0);

  hipMemsetAsync(deg, 0, sizeof(float) * NI, stream);
  deg_count_kernel<<<cdiv(E_II, T), T, 0, stream>>>(ii_src, deg, E_II, 0);
  deg_to_dinv_kernel<<<cdiv(NI, T), T, 0, stream>>>(deg, NI);
  edge_vals_kernel<<<cdiv(E_II, T), T, 0, stream>>>(ii_src, ii_dst, deg, iiVal, E_II, 0, 0);

  hipMemsetAsync(deg, 0, sizeof(float) * NALL, stream);
  deg_count_kernel<<<cdiv(E_UI, T), T, 0, stream>>>(ui_u, deg, E_UI, 0);
  deg_count_kernel<<<cdiv(E_UI, T), T, 0, stream>>>(ui_i, deg, E_UI, NU);
  deg_to_dinv_kernel<<<cdiv(NALL, T), T, 0, stream>>>(deg, NALL);
  edge_vals_kernel<<<cdiv(E_UI, T), T, 0, stream>>>(ui_u, ui_i, deg, uiVal, E_UI, 0, NU);

  // ---- 2. gated embeddings (WMMA + sigmoid-gate epilogue) ---------------
  wmma_gemm_t<DD, float, float><<<gemm_blocks(NU, DD), T, 0, stream>>>(
      user_emb, DD, gwuT, gwub, user_emb, curU, DD, NU, DD, 2);
  wmma_gemm_t<DD, float, float><<<gemm_blocks(NI, DD), T, 0, stream>>>(
      item_emb, DD, gwiT, gwib, item_emb, curI, DD, NI, DD, 2);

  // acc init: accU=uu0, accI=ii0, accUI=concat(raw emb); curUI=concat(raw emb)
  hipMemcpyAsync(accU, curU, sizeof(float) * (size_t)NU * DD, hipMemcpyDeviceToDevice, stream);
  hipMemcpyAsync(accI, curI, sizeof(float) * (size_t)NI * DD, hipMemcpyDeviceToDevice, stream);
  hipMemcpyAsync(curUI, user_emb, sizeof(float) * (size_t)NU * DD, hipMemcpyDeviceToDevice, stream);
  hipMemcpyAsync(curUI + (size_t)NU * DD, item_emb, sizeof(float) * (size_t)NI * DD, hipMemcpyDeviceToDevice, stream);
  hipMemcpyAsync(accUI, user_emb, sizeof(float) * (size_t)NU * DD, hipMemcpyDeviceToDevice, stream);
  hipMemcpyAsync(accUI + (size_t)NU * DD, item_emb, sizeof(float) * (size_t)NI * DD, hipMemcpyDeviceToDevice, stream);

  // ---- 3. propagation layers --------------------------------------------
  for (int l = 0; l < LLAY; ++l) {
    hipMemsetAsync(tmpU, 0, sizeof(float) * (size_t)2 * NALL * DD, stream);
    spmm_atomic_kernel<<<cdiv(E_UU * 32, T), T, 0, stream>>>(uu_src, uu_dst, uuVal, curU, tmpU, E_UU);
    spmm_atomic_kernel<<<cdiv(E_II * 32, T), T, 0, stream>>>(ii_src, ii_dst, iiVal, curI, tmpI, E_II);
    spmm_ui_kernel<<<cdiv(E_UI * 32, T), T, 0, stream>>>(ui_u, ui_i, uiVal, curUI, tmpUI, E_UI);
    layer_combine_kernel<<<cdiv(NALL * 32, T), T, 0, stream>>>(
        tmpU, tmpI, tmpUI, curU, curI, curUI, accU, accI, accUI);
  }

  // ---- 4. means ----------------------------------------------------------
  const float invL = 1.0f / (LLAY + 1);
  scale_inplace_kernel<<<cdiv(NU * DD, T), T, 0, stream>>>(accU, NU * DD, invL);  // userE
  scale_inplace_kernel<<<cdiv(NI * DD, T), T, 0, stream>>>(accI, NI * DD, invL);  // itemE
  finalize_ui_kernel<<<cdiv(NALL * DD, T), T, 0, stream>>>(accUI, out + oUiU, out + oUiI);

  // ---- 5. neighbor aggregation over u-i graph (ones) --------------------
  hipMemsetAsync(tmpU, 0, sizeof(float) * (size_t)NU * DD, stream);
  hipMemsetAsync(tmpI, 0, sizeof(float) * (size_t)NI * DD, stream);
  spmm_atomic_kernel<<<cdiv(E_UI * 32, T), T, 0, stream>>>(ui_u, ui_i, nullptr, out + oUiI, tmpU, E_UI);
  spmm_atomic_kernel<<<cdiv(E_UI * 32, T), T, 0, stream>>>(ui_i, ui_u, nullptr, out + oUiU, tmpI, E_UI);

  // ---- 6. tu / ti projections (concat-3 WMMA, K=192) --------------------
  wmma_gemm_cat3_kernel<<<gemm_blocks(NU, DD), T, 0, stream>>>(
      accU, out + oUiU, tmpU, mnuT, mnu_b, tu, NU);
  wmma_gemm_cat3_kernel<<<gemm_blocks(NI, DD), T, 0, stream>>>(
      accI, out + oUiI, tmpI, mniT, mni_b, ti, NI);

  // ---- 7. MLP heads + softmax einsums -----------------------------------
  auto run_head = [&](const float* x, int n, const float* Ea,
                      int w1i, int w2i, float* dst) {
    // m1 = l2norm(leaky(x@Wp+bp)@Wo+bo); hidden h stored as f16
    wmma_gemm_t<DD, float, _Float16><<<gemm_blocks(n, DK), T, 0, stream>>>(
        x, DD, WpT[w1i], mlpW[w1i][1], nullptr, hbuf, DK, n, DK, 1);
    wmma_gemm_t<DK, _Float16, float><<<gemm_blocks(n, DK), T, 0, stream>>>(
        hbuf, DK, WoT[w1i], mlpW[w1i][3], nullptr, m1, DK, n, DK, 0);
    l2norm_rows_kernel<<<cdiv(n * 32, T), T, 0, stream>>>(m1, m1, n, DK);
    // m2
    wmma_gemm_t<DD, float, _Float16><<<gemm_blocks(n, DK), T, 0, stream>>>(
        x, DD, WpT[w2i], mlpW[w2i][1], nullptr, hbuf, DK, n, DK, 1);
    wmma_gemm_t<DK, _Float16, float><<<gemm_blocks(n, DK), T, 0, stream>>>(
        hbuf, DK, WoT[w2i], mlpW[w2i][3], nullptr, m2, DK, n, DK, 0);
    l2norm_rows_kernel<<<cdiv(n * 32, T), T, 0, stream>>>(m2, m2, n, DK);
    // column means, softmaxes, einsums, residual add
    col_mean_kernel<<<DK, T, 0, stream>>>(m1, n, mean1);
    col_mean_kernel<<<DK, T, 0, stream>>>(m2, n, mean2);
    softmax_d_dot_kernel<<<cdiv(n * KK, T), T, 0, stream>>>(m1, mean1, Ea, sbuf, n);
    softmax_k_out_kernel<<<cdiv(n * DD, T), T, 0, stream>>>(m2, mean2, sbuf, Ea, dst, n);
  };

  run_head(tu, NU, accU, 0, 1, out + oUser);
  run_head(ti, NI, accI, 2, 3, out + oItem);
}
